// SubMAttention3d_2972117369414
// MI455X (gfx1250) — matrix operand order repaired and checked by reading, not compiled, not run on hardware
//
#include <hip/hip_runtime.h>

#define M_VOX 20000
#define K_NBR 32
#define C_DIM 128
#define H_HEADS 8
#define D_HEAD 16
#define FF_DIM 256

typedef __attribute__((ext_vector_type(4)))  unsigned int u32x4;
typedef __attribute__((ext_vector_type(16))) __bf16       v16bf;
typedef __attribute__((ext_vector_type(8)))  float        v8f;

struct FragBits { u32x4 lo, hi; };

static __device__ __forceinline__ v16bf make_frag(u32x4 lo, u32x4 hi) {
  FragBits f{lo, hi};
  return __builtin_bit_cast(v16bf, f);
}

static __device__ __forceinline__ unsigned short f2bf(float f) {
  unsigned int u = __float_as_uint(f);
  u += 0x7FFFu + ((u >> 16) & 1u);   // round-to-nearest-even
  return (unsigned short)(u >> 16);
}

// A fragment (16x32 bf16), row-major source (stride in ushorts applied by caller).
// ISA layout: lane<16 holds K {0..7, 16..23}; lane>=16 holds K {8..15, 24..31}
// of row (lane&15). Two 16B LDS loads per fragment.
static __device__ __forceinline__ v16bf load_a(const unsigned short* base, int hi8) {
  u32x4 lo = *(const u32x4*)(base + hi8);
  u32x4 hi = *(const u32x4*)(base + 16 + hi8);
  return make_frag(lo, hi);
}

// B fragment (32x16 bf16) read from row-major W[n][k] (i.e. B transposed in mem):
// lane has fixed N=(lane&15); lane<16 holds K {0..7, 16..23}, lane>=16 holds K {8..15, 24..31}.
static __device__ __forceinline__ v16bf load_b(const unsigned short* brow, int bko) {
  u32x4 lo = *(const u32x4*)(brow + bko);
  u32x4 hi = *(const u32x4*)(brow + bko + 16);
  return make_frag(lo, hi);
}

static __device__ __forceinline__ v8f wmma_bf16(v16bf a, v16bf b, v8f c) {
  return __builtin_amdgcn_wmma_f32_16x16x32_bf16(
      /*neg_a=*/false, a, /*neg_b=*/false, b,
      /*c_mod=*/(short)0, c, /*reuse_a=*/false, /*reuse_b=*/false);
}

// ---------------------------------------------------------------- prep: bf16 weights
__global__ __launch_bounds__(256) void prep_weights(
    const float* __restrict__ in_proj_w, const float* __restrict__ w1,
    const float* __restrict__ w2,
    unsigned short* __restrict__ wkb, unsigned short* __restrict__ wvb,
    unsigned short* __restrict__ w1b, unsigned short* __restrict__ w2b) {
  int i = blockIdx.x * 256 + threadIdx.x;
  if (i < C_DIM * C_DIM) {
    wkb[i] = f2bf(4.0f * in_proj_w[C_DIM * C_DIM + i]);  // fold SCALE=sqrt(D) into wk
    wvb[i] = f2bf(in_proj_w[2 * C_DIM * C_DIM + i]);
  }
  if (i < FF_DIM * C_DIM) {
    w1b[i] = f2bf(w1[i]);
    w2b[i] = f2bf(w2[i]);
  }
}

// ---------------------------------------------------------------- attention + LN1
__global__ __launch_bounds__(128) void attn_kernel(
    const float* __restrict__ vf, const int* __restrict__ indices,
    const long long* __restrict__ key_idx, const unsigned char* __restrict__ key_mask,
    const float* __restrict__ in_proj_w, const float* __restrict__ in_proj_b,
    const float* __restrict__ out_w, const float* __restrict__ out_b,
    const float* __restrict__ pos_w, const float* __restrict__ pos_b,
    const float* __restrict__ ln1_g, const float* __restrict__ ln1_b,
    const unsigned short* __restrict__ wkb, const unsigned short* __restrict__ wvb,
    float* __restrict__ x1_out) {
  __shared__ __align__(16) unsigned short kf_bf[K_NBR * C_DIM];  // gathered keys (bf16)
  __shared__ __align__(16) float kmat[K_NBR * C_DIM];            // K projection
  __shared__ __align__(16) float vmat[K_NBR * C_DIM];            // V projection
  __shared__ float vfm[C_DIM], qs[C_DIM], ctxs[C_DIM], xs[C_DIM];
  __shared__ float kcoord[K_NBR * 3];
  __shared__ int   nbr[K_NBR];
  __shared__ float sc[H_HEADS * K_NBR];
  __shared__ float smax[H_HEADS], sinv[H_HEADS];
  __shared__ float red[2];

  const int m    = blockIdx.x;
  const int tid  = threadIdx.x;
  const int lane = tid & 31;
  const int wave = tid >> 5;

  // neighbor ids + relative coords (indices[:, [3,2,1]] * 0.1 + min_range)
  if (tid < K_NBR) {
    long long n = key_idx[(long long)m * K_NBR + tid];
    nbr[tid] = (int)n;
    const int* im = indices + m * 4;
    const int* in_ = indices + (int)n * 4;
    kcoord[tid * 3 + 0] = ((float)in_[3] - (float)im[3]) * 0.1f;
    kcoord[tid * 3 + 1] = ((float)in_[2] - (float)im[2]) * 0.1f;
    kcoord[tid * 3 + 2] = ((float)in_[1] - (float)im[1]) * 0.1f;
  }
  vfm[tid] = vf[(long long)m * C_DIM + tid];
  __syncthreads();

  // key_feat = gather(vf) + relu(pos_emb)  -> bf16 A matrix in LDS
  {
    const int c = tid;
    const float pw0 = pos_w[c * 3 + 0], pw1 = pos_w[c * 3 + 1], pw2 = pos_w[c * 3 + 2];
    const float pb = pos_b[c];
    for (int k = 0; k < K_NBR; ++k) {
      float pe = pb + kcoord[k * 3 + 0] * pw0 + kcoord[k * 3 + 1] * pw1 +
                 kcoord[k * 3 + 2] * pw2;
      pe = fmaxf(pe, 0.0f);
      float x = vf[(long long)nbr[k] * C_DIM + c] + pe;
      kf_bf[k * C_DIM + c] = f2bf(x);
    }
  }
  // q = SCALE * vf[m] @ wq^T + bq  (f32 VALU, one output channel per thread)
  {
    const int c = tid;
    const float* wr = in_proj_w + c * C_DIM;
    float acc = 0.f;
    #pragma unroll 8
    for (int i = 0; i < C_DIM; ++i) acc += vfm[i] * wr[i];
    qs[c] = 4.0f * acc + in_proj_b[c];
  }
  __syncthreads();

  // K/V projections via WMMA: 2 matrices x (2 M-tiles x 8 N-tiles) x 4 k-chunks
  {
    const int hi8 = (lane >= 16) ? 8 : 0;
    const int bko = (lane >= 16) ? 8 : 0;
    const int nlo = lane & 15;
    for (int t = wave; t < 32; t += 4) {
      const int mat = t >> 4;       // 0 = K, 1 = V
      const int mt  = (t >> 3) & 1; // key-row tile
      const int nt  = t & 7;        // output-channel tile
      const int nbase = nt * 16;
      const unsigned short* Wb = (mat ? wvb : wkb) + (nbase + nlo) * C_DIM;
      const unsigned short* Ab = kf_bf + (mt * 16 + nlo) * C_DIM;
      v8f acc = {};
      #pragma unroll
      for (int kb = 0; kb < 4; ++kb) {
        v16bf a = load_a(Ab + kb * 32, hi8);
        v16bf b = load_b(Wb + kb * 32, bko);
        acc = wmma_bf16(a, b, acc);
      }
      const int col = nbase + nlo;
      const float bias = in_proj_b[C_DIM + mat * C_DIM + col];
      float* dst = mat ? vmat : kmat;
      #pragma unroll
      for (int j = 0; j < 8; ++j) {
        int row = mt * 16 + j + ((lane >= 16) ? 8 : 0);
        dst[row * C_DIM + col] = acc[j] + bias;
      }
    }
  }
  __syncthreads();

  // scores (mask folded), per-head softmax over K=32
  {
    const unsigned char* msk = key_mask + (long long)m * K_NBR;
    #pragma unroll
    for (int it = 0; it < 2; ++it) {
      int i = tid + it * 128;
      int h = i >> 5, k = i & 31;
      float s = -1.0e9f;
      if (!msk[k]) {
        s = 0.f;
        const float* qp = qs + h * D_HEAD;
        const float* kp = kmat + k * C_DIM + h * D_HEAD;
        #pragma unroll
        for (int d = 0; d < D_HEAD; ++d) s += qp[d] * kp[d];
        s *= 0.25f;  // 1/sqrt(D)
      }
      sc[i] = s;
    }
  }
  __syncthreads();
  if (tid < H_HEADS) {
    float mx = -3.4e38f;
    for (int k = 0; k < K_NBR; ++k) mx = fmaxf(mx, sc[tid * K_NBR + k]);
    float sum = 0.f;
    for (int k = 0; k < K_NBR; ++k) sum += __expf(sc[tid * K_NBR + k] - mx);
    smax[tid] = mx;
    sinv[tid] = 1.0f / sum;
  }
  __syncthreads();
  #pragma unroll
  for (int it = 0; it < 2; ++it) {
    int i = tid + it * 128;
    int h = i >> 5;
    sc[i] = __expf(sc[i] - smax[h]) * sinv[h];
  }
  __syncthreads();

  // ctx = attn @ V   (one channel per thread)
  {
    const int c = tid, h = c >> 4;
    float acc = 0.f;
    #pragma unroll 8
    for (int k = 0; k < K_NBR; ++k) acc += sc[h * K_NBR + k] * vmat[k * C_DIM + c];
    ctxs[c] = acc;
  }
  __syncthreads();
  // out proj + residual
  {
    const int c = tid;
    const float* wr = out_w + c * C_DIM;
    float acc = 0.f;
    #pragma unroll 8
    for (int i = 0; i < C_DIM; ++i) acc += ctxs[i] * wr[i];
    xs[c] = vfm[c] + acc + out_b[c];
  }
  __syncthreads();
  if (tid == 0) {
    float mu = 0.f;
    for (int i = 0; i < C_DIM; ++i) mu += xs[i];
    mu *= (1.0f / C_DIM);
    float var = 0.f;
    for (int i = 0; i < C_DIM; ++i) { float d = xs[i] - mu; var += d * d; }
    var *= (1.0f / C_DIM);
    red[0] = mu;
    red[1] = rsqrtf(var + 1e-5f);
  }
  __syncthreads();
  {
    const int c = tid;
    x1_out[(long long)m * C_DIM + c] = (xs[c] - red[0]) * red[1] * ln1_g[c] + ln1_b[c];
  }
}

// ---------------------------------------------------------------- FFN + LN2 (in-place over x1)
__global__ __launch_bounds__(128) void ffn_kernel(
    const float* __restrict__ b1, const float* __restrict__ b2,
    const float* __restrict__ ln2_g, const float* __restrict__ ln2_b,
    const unsigned short* __restrict__ w1b, const unsigned short* __restrict__ w2b,
    float* __restrict__ x /* in-out: x1 -> final */) {
  __shared__ __align__(16) float          xres[16 * C_DIM];
  __shared__ __align__(16) unsigned short xbf[16 * C_DIM];
  __shared__ __align__(16) unsigned short hbf[16 * FF_DIM];
  __shared__ __align__(16) float          x2[16 * C_DIM];
  __shared__ float mu[16], rs[16];

  const int r0   = blockIdx.x * 16;
  const int tid  = threadIdx.x;
  const int lane = tid & 31;
  const int wave = tid >> 5;
  const int hi8  = (lane >= 16) ? 8 : 0;
  const int bko  = (lane >= 16) ? 8 : 0;
  const int nlo  = lane & 15;

  // stage x1 tile (residual f32 + A-matrix bf16)
  #pragma unroll
  for (int r = 0; r < 16; ++r) {
    float v = x[(long long)(r0 + r) * C_DIM + tid];
    xres[r * C_DIM + tid] = v;
    xbf[r * C_DIM + tid] = f2bf(v);
  }
  __syncthreads();

  // GEMM1: (16 x 128) @ w1^T -> (16 x 256), ReLU, bf16 to LDS
  for (int nt = wave; nt < 16; nt += 4) {
    const int nbase = nt * 16;
    const unsigned short* Wb = w1b + (nbase + nlo) * C_DIM;
    const unsigned short* Ab = xbf + nlo * C_DIM;
    v8f acc = {};
    #pragma unroll
    for (int kb = 0; kb < 4; ++kb) {
      v16bf a = load_a(Ab + kb * 32, hi8);
      v16bf b = load_b(Wb + kb * 32, bko);
      acc = wmma_bf16(a, b, acc);
    }
    const int col = nbase + nlo;
    const float bias = b1[col];
    #pragma unroll
    for (int j = 0; j < 8; ++j) {
      int row = j + ((lane >= 16) ? 8 : 0);
      hbf[row * FF_DIM + col] = f2bf(fmaxf(acc[j] + bias, 0.0f));
    }
  }
  __syncthreads();

  // GEMM2: (16 x 256) @ w2^T -> (16 x 128), + bias + residual
  for (int nt = wave; nt < 8; nt += 4) {
    const int nbase = nt * 16;
    const unsigned short* Wb = w2b + (nbase + nlo) * FF_DIM;
    const unsigned short* Ab = hbf + nlo * FF_DIM;
    v8f acc = {};
    #pragma unroll
    for (int kb = 0; kb < 8; ++kb) {
      v16bf a = load_a(Ab + kb * 32, hi8);
      v16bf b = load_b(Wb + kb * 32, bko);
      acc = wmma_bf16(a, b, acc);
    }
    const int col = nbase + nlo;
    const float bias = b2[col];
    #pragma unroll
    for (int j = 0; j < 8; ++j) {
      int row = j + ((lane >= 16) ? 8 : 0);
      x2[row * C_DIM + col] = xres[row * C_DIM + col] + acc[j] + bias;
    }
  }
  __syncthreads();

  if (tid < 16) {
    const float* xr = x2 + tid * C_DIM;
    float m_ = 0.f;
    for (int i = 0; i < C_DIM; ++i) m_ += xr[i];
    m_ *= (1.0f / C_DIM);
    float var = 0.f;
    for (int i = 0; i < C_DIM; ++i) { float d = xr[i] - m_; var += d * d; }
    var *= (1.0f / C_DIM);
    mu[tid] = m_;
    rs[tid] = rsqrtf(var + 1e-5f);
  }
  __syncthreads();
  #pragma unroll
  for (int r = 0; r < 16; ++r) {
    x[(long long)(r0 + r) * C_DIM + tid] =
        (x2[r * C_DIM + tid] - mu[r]) * rs[r] * ln2_g[tid] + ln2_b[tid];
  }
}

extern "C" void kernel_launch(void* const* d_in, const int* in_sizes, int n_in,
                              void* d_out, int out_size, void* d_ws, size_t ws_size,
                              hipStream_t stream) {
  (void)in_sizes; (void)n_in; (void)out_size; (void)ws_size;
  const float*         vf        = (const float*)d_in[0];
  const int*           indices   = (const int*)d_in[1];
  const long long*     key_idx   = (const long long*)d_in[2];   // jnp.int64
  const unsigned char* key_mask  = (const unsigned char*)d_in[3];
  const float*         in_proj_w = (const float*)d_in[4];
  const float*         in_proj_b = (const float*)d_in[5];
  const float*         out_w     = (const float*)d_in[6];
  const float*         out_b     = (const float*)d_in[7];
  const float*         pos_w     = (const float*)d_in[8];
  const float*         pos_b     = (const float*)d_in[9];
  const float*         w1        = (const float*)d_in[10];
  const float*         b1        = (const float*)d_in[11];
  const float*         w2        = (const float*)d_in[12];
  const float*         b2        = (const float*)d_in[13];
  const float*         ln1_g     = (const float*)d_in[14];
  const float*         ln1_b     = (const float*)d_in[15];
  const float*         ln2_g     = (const float*)d_in[16];
  const float*         ln2_b     = (const float*)d_in[17];
  float* out = (float*)d_out;

  unsigned short* wkb = (unsigned short*)d_ws;              // 128*128
  unsigned short* wvb = wkb + C_DIM * C_DIM;                // 128*128
  unsigned short* w1b = wvb + C_DIM * C_DIM;                // 256*128
  unsigned short* w2b = w1b + FF_DIM * C_DIM;               // 128*256

  prep_weights<<<(FF_DIM * C_DIM + 255) / 256, 256, 0, stream>>>(
      in_proj_w, w1, w2, wkb, wvb, w1b, w2b);

  attn_kernel<<<M_VOX, 128, 0, stream>>>(
      vf, indices, key_idx, key_mask, in_proj_w, in_proj_b, out_w, out_b,
      pos_w, pos_b, ln1_g, ln1_b, wkb, wvb, out);

  ffn_kernel<<<M_VOX / 16, 128, 0, stream>>>(
      b1, b2, ln2_g, ln2_b, w1b, w2b, out);
}
